// ScaledDotProductAttention_51591147159989
// MI455X (gfx1250) — matrix operand order repaired and checked by reading, not compiled, not run on hardware
//
#include <hip/hip_runtime.h>
#include <hip/hip_bf16.h>

// MI455X / gfx1250 relative-position attention (Transformer-XL style)
// out = softmax(((q+uT)K^T + shift((q+vT)R^T)) / sqrt(64)) V
// Shift identity: BD'[qi,kj] = BD[qi, kj - qi + 1023] if kj-qi <= 1024 else 0.
// BD computed as a banded WMMA GEMM per k-tile + diagonal LDS gather.

typedef __attribute__((ext_vector_type(16))) __bf16 v16bf;
typedef __attribute__((ext_vector_type(4)))  __bf16 v4bf;
typedef __attribute__((ext_vector_type(8)))  float  v8f;

namespace {
constexpr int kB = 2, kH = 16, kQ = 1024, kK = 2048, kD = 64;
constexpr int KT = 64;            // k-tile width per iteration
constexpr int BAND = 80;          // band width (79 needed, round to 5 subtiles)
constexpr float kScale = 0.125f;  // 1/sqrt(64)
}

#define WMMA_BF16(a, b, c) \
  __builtin_amdgcn_wmma_f32_16x16x32_bf16(false, (a), false, (b), (short)0, (c), false, false)

__device__ inline v4bf cvt4(float4 f) {
  v4bf o;
  o[0] = (__bf16)f.x; o[1] = (__bf16)f.y;
  o[2] = (__bf16)f.z; o[3] = (__bf16)f.w;
  return o;
}

// A-fragment (16x32 bf16, M x Kdepth) from row-major LDS tile, row stride 64.
// Layout: lane m = lane&15, hi = lane>>4; VGPR i holds K pair
// K0 = 2*i + 16*hi (contiguous, matches B-fragment).
__device__ inline v16bf load_a_frag(const __bf16* tile, int kOff, int m16, int hi) {
  const __bf16* row = tile + m16 * 64 + kOff + 16 * hi;
  v16bf a;
#pragma unroll
  for (int i = 0; i < 8; ++i) {
    const int K0 = 2 * i;
    a[2 * i]     = row[K0];
    a[2 * i + 1] = row[K0 + 1];
  }
  return a;
}

// B-fragment (32x16 bf16, Kdepth x N). tile is [n][kdepth] row-major, stride 64.
// Layout: lanes 0-15 hold K=0..15, lanes 16-31 hold K=16..31, N = lane&15,
// K contiguous per lane -> 32B vector load (alignas(32) LDS, 32B offsets).
__device__ inline v16bf load_b_frag(const __bf16* tile, int n0, int kOff, int m16, int hi) {
  const __bf16* p = tile + (n0 + m16) * 64 + kOff + 16 * hi;
  return *(const v16bf*)p;
}

__global__ __launch_bounds__(128, 1) void relattn_fwd_gfx1250(
    const float* __restrict__ q, const float* __restrict__ k,
    const float* __restrict__ v, const float* __restrict__ r,
    const float* __restrict__ uT, const float* __restrict__ vT,
    float* __restrict__ out) {
  __shared__ alignas(32) __bf16 sAU[16 * 64];   // (q+uT) tile, bf16
  __shared__ alignas(32) __bf16 sAV[16 * 64];   // (q+vT) tile, bf16
  __shared__ alignas(32) __bf16 sKt[KT * 64];   // k tile   [k_local][d]
  __shared__ alignas(32) __bf16 sVT[64 * KT];   // v tile^T [d][k_local]
  __shared__ alignas(32) __bf16 sR [BAND * 64]; // r band   [band_row][d]
  __shared__ alignas(32) __bf16 sP [16 * KT];   // probabilities, bf16
  __shared__ float sBand[16 * BAND];            // BD band in f32
  __shared__ float sPartMax[4][16];
  __shared__ float sPartSum[4][16];

  const int qt = blockIdx.x;       // q tile (0..63)
  const int h  = blockIdx.y;       // head
  const int b  = blockIdx.z;       // batch
  const int q0 = qt * 16;

  const int tid  = threadIdx.x;
  const int lane = tid & 31;
  const int wv   = tid >> 5;       // wave 0..3 (owns k-cols / d-cols 16*wv..)
  const int m16  = lane & 15;
  const int hi   = lane >> 4;

  const float* qbase = q + (((size_t)b * kH + h) * kQ + q0) * kD;
  const float* kbase = k + (((size_t)b * kH + h) * kK) * kD;
  const float* vbase = v + (((size_t)b * kH + h) * kK) * kD;
  const float* rbase = r + ((size_t)h * kK) * kD;
  const float* ub    = uT + (size_t)h * kD;
  const float* vb    = vT + (size_t)h * kD;

  // Stage (q+uT), (q+vT) once, vectorized float4 -> packed 4xbf16.
  for (int idx = tid; idx < 16 * 16; idx += 128) {
    const int row = idx >> 4, d4 = (idx & 15) * 4;
    const float4 qv = *(const float4*)(qbase + (size_t)row * kD + d4);
    const float4 uv = *(const float4*)(ub + d4);
    const float4 vvv = *(const float4*)(vb + d4);
    v4bf au, av;
    au[0] = (__bf16)(qv.x + uv.x);  av[0] = (__bf16)(qv.x + vvv.x);
    au[1] = (__bf16)(qv.y + uv.y);  av[1] = (__bf16)(qv.y + vvv.y);
    au[2] = (__bf16)(qv.z + uv.z);  av[2] = (__bf16)(qv.z + vvv.z);
    au[3] = (__bf16)(qv.w + uv.w);  av[3] = (__bf16)(qv.w + vvv.w);
    *(v4bf*)&sAU[row * 64 + d4] = au;
    *(v4bf*)&sAV[row * 64 + d4] = av;
  }
  __syncthreads();

  // Loop-invariant A-fragments hoisted to registers (barriers would otherwise
  // force the LDS gathers to repeat every iteration).
  const v16bf aU0 = load_a_frag(sAU, 0,  m16, hi);
  const v16bf aU1 = load_a_frag(sAU, 32, m16, hi);
  const v16bf aV0 = load_a_frag(sAV, 0,  m16, hi);
  const v16bf aV1 = load_a_frag(sAV, 32, m16, hi);

  v8f o = {0.f, 0.f, 0.f, 0.f, 0.f, 0.f, 0.f, 0.f};
  float mrow[8], lrow[8];
#pragma unroll
  for (int j = 0; j < 8; ++j) { mrow[j] = -1.0e30f; lrow[j] = 0.0f; }

  for (int k0 = 0; k0 < kK; k0 += KT) {
    __syncthreads();  // previous iteration's PV reads done before re-staging

    // ---- stage k (row-major) and v (transposed), float4 global loads ----
    for (int idx = tid; idx < KT * 16; idx += 128) {
      const int row = idx >> 4, d4 = (idx & 15) * 4;
      const float4 kv = *(const float4*)(kbase + (size_t)(k0 + row) * kD + d4);
      const float4 vv = *(const float4*)(vbase + (size_t)(k0 + row) * kD + d4);
      *(v4bf*)&sKt[row * 64 + d4] = cvt4(kv);
      sVT[(d4 + 0) * KT + row] = (__bf16)vv.x;
      sVT[(d4 + 1) * KT + row] = (__bf16)vv.y;
      sVT[(d4 + 2) * KT + row] = (__bf16)vv.z;
      sVT[(d4 + 3) * KT + row] = (__bf16)vv.w;
    }
    const int jb = k0 - q0 + 1008;  // band base, always >= 0
    for (int idx = tid; idx < BAND * 16; idx += 128) {
      const int row = idx >> 4, d4 = (idx & 15) * 4;
      const int jp = jb + row;
      v4bf rv;
      if (jp < kK) {
        rv = cvt4(*(const float4*)(rbase + (size_t)jp * kD + d4));
      } else {
        rv[0] = rv[1] = rv[2] = rv[3] = (__bf16)0.0f;
      }
      *(v4bf*)&sR[row * 64 + d4] = rv;
    }

    // ---- prefetch next k-tile (global_prefetch_b8): overlap next tile's
    // L2->WGP fill with this iteration's WMMA + softmax work.
    if (k0 + KT < kK) {
      const float* nk = kbase + (size_t)(k0 + KT) * kD;
      const float* nv = vbase + (size_t)(k0 + KT) * kD;
      __builtin_prefetch(nk + (size_t)tid * 32, 0, 1);  // one 128B line each
      __builtin_prefetch(nv + (size_t)tid * 32, 0, 1);
      const int jb2 = k0 + KT - q0 + 1008;
      if (tid < BAND && jb2 + tid < kK) {
        const float* nr = rbase + (size_t)(jb2 + tid) * kD;
        __builtin_prefetch(nr, 0, 1);
        __builtin_prefetch(nr + 32, 0, 1);
      }
    }
    __syncthreads();

    // ---- BD band GEMM: (q+vT)[16x64] @ r[band]^T -> 16x80 (5 subtiles) ----
    for (int s = wv; s < 5; s += 4) {  // wave-uniform branch: EXEC stays full
      v8f c = {0.f, 0.f, 0.f, 0.f, 0.f, 0.f, 0.f, 0.f};
      v16bf bb = load_b_frag(sR, 16 * s, 0, m16, hi);
      c = WMMA_BF16(aV0, bb, c);
      bb = load_b_frag(sR, 16 * s, 32, m16, hi);
      c = WMMA_BF16(aV1, bb, c);
#pragma unroll
      for (int j = 0; j < 8; ++j)
        sBand[(j + 8 * hi) * BAND + 16 * s + m16] = c[j];
    }

    // ---- AC GEMM: (q+uT)[16x64] @ k^T -> each wave one 16x16 subtile ----
    v8f sc = {0.f, 0.f, 0.f, 0.f, 0.f, 0.f, 0.f, 0.f};
    {
      v16bf bb = load_b_frag(sKt, 16 * wv, 0, m16, hi);
      sc = WMMA_BF16(aU0, bb, sc);
      bb = load_b_frag(sKt, 16 * wv, 32, m16, hi);
      sc = WMMA_BF16(aU1, bb, sc);
    }
    __syncthreads();  // band visible

    // ---- S = (AC + shifted BD) * scale ----
    float sv[8];
#pragma unroll
    for (int j = 0; j < 8; ++j) {
      const int row  = j + 8 * hi;            // local q row
      const int col  = 16 * wv + m16;         // local k col
      const int bcol = col - row + 15;        // diagonal gather, in [0,79)
      sv[j] = (sc[j] + sBand[row * BAND + bcol]) * kScale;
    }

    // ---- online softmax: per-row max over this k-tile ----
    float tm[8];
#pragma unroll
    for (int j = 0; j < 8; ++j) {
      float t = sv[j];
      t = fmaxf(t, __shfl_xor(t, 1, 32));
      t = fmaxf(t, __shfl_xor(t, 2, 32));
      t = fmaxf(t, __shfl_xor(t, 4, 32));
      t = fmaxf(t, __shfl_xor(t, 8, 32));
      tm[j] = t;
    }
    if (m16 == 0) {
#pragma unroll
      for (int j = 0; j < 8; ++j) sPartMax[wv][j + 8 * hi] = tm[j];
    }
    __syncthreads();

    float mnew[8], esc[8], p[8], ts[8];
#pragma unroll
    for (int j = 0; j < 8; ++j) {
      const int row = j + 8 * hi;
      const float rm = fmaxf(fmaxf(sPartMax[0][row], sPartMax[1][row]),
                             fmaxf(sPartMax[2][row], sPartMax[3][row]));
      mnew[j] = fmaxf(mrow[j], rm);
      esc[j]  = __expf(mrow[j] - mnew[j]);
      p[j]    = __expf(sv[j] - mnew[j]);
    }
#pragma unroll
    for (int j = 0; j < 8; ++j) {
      float t = p[j];
      t += __shfl_xor(t, 1, 32);
      t += __shfl_xor(t, 2, 32);
      t += __shfl_xor(t, 4, 32);
      t += __shfl_xor(t, 8, 32);
      ts[j] = t;
    }
    if (m16 == 0) {
#pragma unroll
      for (int j = 0; j < 8; ++j) sPartSum[wv][j + 8 * hi] = ts[j];
    }
#pragma unroll
    for (int j = 0; j < 8; ++j)
      sP[(j + 8 * hi) * KT + 16 * wv + m16] = (__bf16)p[j];
    __syncthreads();  // sP + sPartSum visible

#pragma unroll
    for (int j = 0; j < 8; ++j) {
      const int row = j + 8 * hi;
      const float rs = sPartSum[0][row] + sPartSum[1][row] +
                       sPartSum[2][row] + sPartSum[3][row];
      lrow[j] = lrow[j] * esc[j] + rs;
      mrow[j] = mnew[j];
      o[j] *= esc[j];
    }

    // ---- PV GEMM: P[16x64] @ V[64x16(d)] accumulated into o ----
    {
      v16bf pa = load_a_frag(sP, 0, m16, hi);
      v16bf bv = load_b_frag(sVT, 16 * wv, 0, m16, hi);
      o = WMMA_BF16(pa, bv, o);
      pa = load_a_frag(sP, 32, m16, hi);
      bv = load_b_frag(sVT, 16 * wv, 32, m16, hi);
      o = WMMA_BF16(pa, bv, o);
    }
  }

  // ---- epilogue: normalize and store f32 output ----
  float* ob = out + (((size_t)b * kH + h) * kQ + q0) * kD;
#pragma unroll
  for (int j = 0; j < 8; ++j) {
    const int row = j + 8 * hi;
    ob[(size_t)row * kD + 16 * wv + m16] = o[j] / lrow[j];
  }
}

extern "C" void kernel_launch(void* const* d_in, const int* in_sizes, int n_in,
                              void* d_out, int out_size, void* d_ws, size_t ws_size,
                              hipStream_t stream) {
  (void)in_sizes; (void)n_in; (void)d_ws; (void)ws_size; (void)out_size;
  const float* q  = (const float*)d_in[0];
  const float* k  = (const float*)d_in[1];
  const float* v  = (const float*)d_in[2];
  const float* r  = (const float*)d_in[3];
  const float* uT = (const float*)d_in[4];
  const float* vT = (const float*)d_in[5];
  // d_in[6] = mask: all-ones in the reference and unused by its math -> ignored.
  float* out = (float*)d_out;

  dim3 grid(kQ / 16, kH, kB);  // 64 q-tiles x 16 heads x 2 batch = 2048 blocks
  relattn_fwd_gfx1250<<<grid, 128, 0, stream>>>(q, k, v, r, uT, vT, out);
}
